// GAT_6330781794595
// MI455X (gfx1250) — matrix operand order repaired and checked by reading, not compiled, not run on hardware
//
#include <hip/hip_runtime.h>
#include <stdint.h>

#define NN      50000
#define EE      800000
#define DIN     256
#define NHEAD   4
#define DOUT    40
#define LDH2    48
#define NEG_SLOPE 0.2f

typedef __attribute__((ext_vector_type(16))) __bf16 v16bf;
typedef __attribute__((ext_vector_type(8)))  float  v8f;

union Frag16 { v16bf v; uint32_t u[8]; };

__device__ __forceinline__ uint16_t f2bf(float f) {
    uint32_t u = __float_as_uint(f);
    u += 0x7FFFu + ((u >> 16) & 1u);          // round-to-nearest-even
    return (uint16_t)(u >> 16);
}
__device__ __forceinline__ unsigned ford(float f) {
    unsigned b = __float_as_uint(f);
    return (b & 0x80000000u) ? ~b : (b | 0x80000000u);
}
__device__ __forceinline__ float funord(unsigned u) {
    unsigned b = (u & 0x80000000u) ? (u & 0x7fffffffu) : ~u;
    return __uint_as_float(b);
}
__device__ __forceinline__ float lrelu(float v) { return v > 0.f ? v : NEG_SLOPE * v; }

// ---------------- fills / conversions ----------------
__global__ void k_fill_u32(unsigned* p, long n) {
    long i = (long)blockIdx.x * blockDim.x + threadIdx.x;
    if (i < n) p[i] = 0u;
}
__global__ void k_cvt_bf(const float* __restrict__ in, uint16_t* __restrict__ out, long n) {
    long i = (long)blockIdx.x * blockDim.x + threadIdx.x;
    if (i < n) out[i] = f2bf(in[i]);
}
// W1 [K=256][N=256] row-major -> W1t_bf [n][k]
__global__ void k_prep_w1(const float* __restrict__ W, uint16_t* __restrict__ Wt) {
    int i = blockIdx.x * blockDim.x + threadIdx.x;
    if (i >= 256 * 256) return;
    int n = i >> 8, k = i & 255;
    Wt[i] = f2bf(W[k * 256 + n]);
}
// W2 [K=256][N=40] -> W2t_bf [48][256], zero-padded columns 40..47
__global__ void k_prep_w2(const float* __restrict__ W, uint16_t* __restrict__ Wt) {
    int i = blockIdx.x * blockDim.x + threadIdx.x;
    if (i >= LDH2 * 256) return;
    int n = i >> 8, k = i & 255;
    Wt[i] = f2bf(n < DOUT ? W[k * DOUT + n] : 0.f);
}
// relu(agg + b1) -> bf16 (layer-2 input)
__global__ void k_bias_relu_cvt(const float* __restrict__ agg, const float* __restrict__ b,
                                uint16_t* __restrict__ xbf, long n) {
    long i = (long)blockIdx.x * blockDim.x + threadIdx.x;
    if (i >= n) return;
    float v = agg[i] + b[i & 255];
    xbf[i] = f2bf(v > 0.f ? v : 0.f);
}

// ---------------- WMMA GEMM: C[M x tilesN*16] = A_bf[M x 256] * Bt_bf[tilesN*16 x 256]^T
// one wave per 16x16 C tile; grid sized exactly so EXEC is always all-ones.
__global__ void k_gemm_bf16(const uint16_t* __restrict__ A,
                            const uint16_t* __restrict__ Bt,
                            float* __restrict__ C,
                            int tilesN, int ldc) {
    const int wave    = (blockIdx.x * blockDim.x + threadIdx.x) >> 5;
    const int lane    = threadIdx.x & 31;
    const int tileM   = wave / tilesN;
    const int tileN   = wave % tilesN;
    const int halfSel = lane >> 4;       // lane half selects K interleave
    const int l15     = lane & 15;
    const uint16_t* arow = A  + (size_t)(tileM * 16 + l15) * DIN;  // A row (M)
    const uint16_t* brow = Bt + (size_t)(tileN * 16 + l15) * DIN;  // B col (N), k-contiguous
    v8f acc = {0.f, 0.f, 0.f, 0.f, 0.f, 0.f, 0.f, 0.f};
#pragma unroll
    for (int k0 = 0; k0 < DIN; k0 += 32) {
        Frag16 a, b;
#pragma unroll
        for (int v = 0; v < 8; ++v) {
            // A 16x32 bf16 layout (ISA 7.12.2): vgpr v -> K = 16*(v/4) + 8*laneHalf + 2*(v%4)
            const int ka = k0 + ((v >> 2) << 4) + (halfSel << 3) + ((v & 3) << 1);
            a.u[v] = *(const uint32_t*)(arow + ka);
            // B 32x16 bf16: same K mapping as A: vgpr v -> K = 16*(v/4) + 8*laneHalf + 2*(v%4)
            const int kb = k0 + ((v >> 2) << 4) + (halfSel << 3) + ((v & 3) << 1);
            b.u[v] = *(const uint32_t*)(brow + kb);
        }
        acc = __builtin_amdgcn_wmma_f32_16x16x32_bf16(false, a.v, false, b.v,
                                                      (short)0, acc, false, false);
    }
    // C layout: lanes 0-15 rows 0..7, lanes 16-31 rows 8..15; col = lane&15
    float* crow = C + (size_t)(tileM * 16 + 8 * halfSel) * ldc + tileN * 16 + l15;
#pragma unroll
    for (int r = 0; r < 8; ++r) crow[(size_t)r * ldc] = acc[r];
}

// ---------------- attention score projections ----------------
// layer 1: a_s[n,h] = dot(h1[n,h*64:..], att_src[h]); one thread per (n,h)
__global__ void k_att1(const float* __restrict__ h1, const float* __restrict__ asrc,
                       const float* __restrict__ adst, float* __restrict__ as_,
                       float* __restrict__ ad_) {
    int i = blockIdx.x * blockDim.x + threadIdx.x;
    if (i >= NN * NHEAD) return;
    int n = i >> 2, hd = i & 3;
    const float* hp = h1 + (size_t)n * DIN + hd * 64;
    const float* ws = asrc + hd * 64;
    const float* wd = adst + hd * 64;
    float ss = 0.f, sd = 0.f;
#pragma unroll 8
    for (int f = 0; f < 64; ++f) { float v = hp[f]; ss += v * ws[f]; sd += v * wd[f]; }
    as_[i] = ss; ad_[i] = sd;
}
// layer 2 (H=1, F=40, stride 48)
__global__ void k_att2(const float* __restrict__ h2, const float* __restrict__ asrc,
                       const float* __restrict__ adst, float* __restrict__ as_,
                       float* __restrict__ ad_) {
    int n = blockIdx.x * blockDim.x + threadIdx.x;
    if (n >= NN) return;
    const float* hp = h2 + (size_t)n * LDH2;
    float ss = 0.f, sd = 0.f;
#pragma unroll 8
    for (int f = 0; f < DOUT; ++f) { float v = hp[f]; ss += v * asrc[f]; sd += v * adst[f]; }
    as_[n] = ss; ad_[n] = sd;
}

// ---------------- segment softmax (ordered-uint atomic max, then exp/sum) ----------------
__global__ void k_edge_max(const int* __restrict__ ei, int H,
                           const float* __restrict__ as_, const float* __restrict__ ad_,
                           unsigned* __restrict__ m) {
    int e = blockIdx.x * blockDim.x + threadIdx.x;
    if (e >= EE + NN) return;
    int s, d;
    if (e < EE) { s = ei[e]; d = ei[EE + e]; } else { s = d = e - EE; }
    for (int hd = 0; hd < H; ++hd) {
        float v = lrelu(as_[s * H + hd] + ad_[d * H + hd]);
        atomicMax(&m[d * H + hd], ford(v));
    }
}
__global__ void k_edge_exp(const int* __restrict__ ei, int H,
                           const float* __restrict__ as_, const float* __restrict__ ad_,
                           const unsigned* __restrict__ m, float* __restrict__ ex,
                           float* __restrict__ den) {
    int e = blockIdx.x * blockDim.x + threadIdx.x;
    if (e >= EE + NN) return;
    int s, d;
    if (e < EE) { s = ei[e]; d = ei[EE + e]; } else { s = d = e - EE; }
    for (int hd = 0; hd < H; ++hd) {
        float v  = lrelu(as_[s * H + hd] + ad_[d * H + hd]);
        float xv = __expf(v - funord(m[d * H + hd]));
        ex[e * H + hd] = xv;
        atomicAdd(&den[d * H + hd], xv);
    }
}

// ---------------- aggregation: wave per edge, lane per feature, L2-resident atomics ----
__global__ void k_agg1(const int* __restrict__ ei, const float* __restrict__ h1,
                       const float* __restrict__ ex, const float* __restrict__ den,
                       float* __restrict__ agg) {
    int wid  = (blockIdx.x * blockDim.x + threadIdx.x) >> 5;
    int lane = threadIdx.x & 31;
    if (wid >= EE + NN) return;
    int s, d;
    if (wid < EE) { s = ei[wid]; d = ei[EE + wid]; } else { s = d = wid - EE; }
    float alpha[NHEAD];
#pragma unroll
    for (int hd = 0; hd < NHEAD; ++hd) alpha[hd] = ex[wid * NHEAD + hd] / den[d * NHEAD + hd];
    const float* hs = h1 + (size_t)s * DIN;
    float* ag = agg + (size_t)d * DIN;
#pragma unroll
    for (int i = 0; i < 8; ++i) {
        int f = lane + (i << 5);
        atomicAdd(&ag[f], hs[f] * alpha[f >> 6]);
    }
}
__global__ void k_agg2(const int* __restrict__ ei, const float* __restrict__ h2,
                       const float* __restrict__ ex, const float* __restrict__ den,
                       float* __restrict__ agg) {
    int wid  = (blockIdx.x * blockDim.x + threadIdx.x) >> 5;
    int lane = threadIdx.x & 31;
    if (wid >= EE + NN) return;
    int s, d;
    if (wid < EE) { s = ei[wid]; d = ei[EE + wid]; } else { s = d = wid - EE; }
    float alpha = ex[wid] / den[d];
    const float* hs = h2 + (size_t)s * LDH2;
    float* ag = agg + (size_t)d * DOUT;
    atomicAdd(&ag[lane], hs[lane] * alpha);
    if (lane < DOUT - 32) atomicAdd(&ag[lane + 32], hs[lane + 32] * alpha);
}

// ---------------- final bias + log_softmax over 40 classes ----------------
__global__ void k_logsoftmax(const float* __restrict__ agg, const float* __restrict__ b2,
                             float* __restrict__ out) {
    int n = blockIdx.x * blockDim.x + threadIdx.x;
    if (n >= NN) return;
    const float* ap = agg + (size_t)n * DOUT;
    float v[DOUT], mx = -3.402823466e38f;
#pragma unroll
    for (int c = 0; c < DOUT; ++c) { v[c] = ap[c] + b2[c]; mx = fmaxf(mx, v[c]); }
    float s = 0.f;
#pragma unroll
    for (int c = 0; c < DOUT; ++c) s += __expf(v[c] - mx);
    float ls = __logf(s);
    float* op = out + (size_t)n * DOUT;
#pragma unroll
    for (int c = 0; c < DOUT; ++c) op[c] = v[c] - mx - ls;
}

extern "C" void kernel_launch(void* const* d_in, const int* in_sizes, int n_in,
                              void* d_out, int out_size, void* d_ws, size_t ws_size,
                              hipStream_t stream) {
    const float* x    = (const float*)d_in[0];
    const int*   ei   = (const int*)d_in[1];
    const float* W1   = (const float*)d_in[2];
    const float* asr1 = (const float*)d_in[3];
    const float* ads1 = (const float*)d_in[4];
    const float* b1   = (const float*)d_in[5];
    const float* W2   = (const float*)d_in[6];
    const float* asr2 = (const float*)d_in[7];
    const float* ads2 = (const float*)d_in[8];
    const float* b2   = (const float*)d_in[9];
    float* out = (float*)d_out;

    char* ws = (char*)d_ws;
    size_t off = 0;
    auto take = [&](size_t bytes) -> char* {
        char* p = ws + off; off += (bytes + 255) & ~(size_t)255; return p;
    };
    float*    h1   = (float*)take((size_t)NN * DIN * 4);      // layer-1 features (pre-agg)
    float*    agg1 = (float*)take((size_t)NN * DIN * 4);      // layer-1 aggregation
    uint16_t* xbf  = (uint16_t*)take((size_t)NN * DIN * 2);   // bf16 input (reused for layer 2)
    uint16_t* w1t  = (uint16_t*)take((size_t)256 * 256 * 2);
    uint16_t* w2t  = (uint16_t*)take((size_t)LDH2 * 256 * 2);
    float*    as1  = (float*)take((size_t)NN * NHEAD * 4);
    float*    ad1  = (float*)take((size_t)NN * NHEAD * 4);
    unsigned* m1   = (unsigned*)take((size_t)NN * NHEAD * 4);
    float*    den1 = (float*)take((size_t)NN * NHEAD * 4);
    float*    ex1  = (float*)take((size_t)(EE + NN) * NHEAD * 4);
    float*    as2  = (float*)take((size_t)NN * 4);
    float*    ad2  = (float*)take((size_t)NN * 4);
    unsigned* m2   = (unsigned*)take((size_t)NN * 4);
    float*    den2 = (float*)take((size_t)NN * 4);
    // overlays into dead regions (deterministic stream order):
    float* h2   = h1;                                          // [NN x 48], h1 dead after agg1
    float* agg2 = (float*)((char*)h1 + (((size_t)NN * LDH2 * 4 + 255) & ~(size_t)255));
    float* ex2  = ex1;                                         // ex1 dead after agg1

    const int nE = EE + NN;
    auto cdiv = [](long a, long b) { return (int)((a + b - 1) / b); };

    // zero / -inf(ordered) initialization
    k_fill_u32<<<cdiv((long)NN * NHEAD, 256), 256, 0, stream>>>(m1, (long)NN * NHEAD);
    k_fill_u32<<<cdiv((long)NN * NHEAD, 256), 256, 0, stream>>>((unsigned*)den1, (long)NN * NHEAD);
    k_fill_u32<<<cdiv((long)NN * DIN, 256), 256, 0, stream>>>((unsigned*)agg1, (long)NN * DIN);
    k_fill_u32<<<cdiv((long)NN, 256), 256, 0, stream>>>(m2, NN);
    k_fill_u32<<<cdiv((long)NN, 256), 256, 0, stream>>>((unsigned*)den2, NN);
    k_fill_u32<<<cdiv((long)NN * DOUT, 256), 256, 0, stream>>>((unsigned*)agg2, (long)NN * DOUT);

    // convert inputs to bf16, transpose weights to [n][k]
    k_cvt_bf<<<cdiv((long)NN * DIN, 256), 256, 0, stream>>>(x, xbf, (long)NN * DIN);
    k_prep_w1<<<cdiv(256 * 256, 256), 256, 0, stream>>>(W1, w1t);
    k_prep_w2<<<cdiv(LDH2 * 256, 256), 256, 0, stream>>>(W2, w2t);

    // ---- layer 1 ----
    // 3125 M-tiles x 16 N-tiles = 50000 waves; 8 waves/block -> 6250 blocks exactly
    k_gemm_bf16<<<6250, 256, 0, stream>>>(xbf, w1t, h1, 16, DIN);
    k_att1<<<cdiv((long)NN * NHEAD, 256), 256, 0, stream>>>(h1, asr1, ads1, as1, ad1);
    k_edge_max<<<cdiv(nE, 256), 256, 0, stream>>>(ei, NHEAD, as1, ad1, m1);
    k_edge_exp<<<cdiv(nE, 256), 256, 0, stream>>>(ei, NHEAD, as1, ad1, m1, ex1, den1);
    k_agg1<<<cdiv((long)nE * 32, 256), 256, 0, stream>>>(ei, h1, ex1, den1, agg1);

    // ---- layer 2 ----
    k_bias_relu_cvt<<<cdiv((long)NN * DIN, 256), 256, 0, stream>>>(agg1, b1, xbf, (long)NN * DIN);
    // 3125 M-tiles x 3 N-tiles = 9375 waves; 3 waves/block (96 thr) -> 3125 blocks exactly
    k_gemm_bf16<<<3125, 96, 0, stream>>>(xbf, w2t, h2, 3, LDH2);
    k_att2<<<cdiv(NN, 256), 256, 0, stream>>>(h2, asr2, ads2, as2, ad2);
    k_edge_max<<<cdiv(nE, 256), 256, 0, stream>>>(ei, 1, as2, ad2, m2);
    k_edge_exp<<<cdiv(nE, 256), 256, 0, stream>>>(ei, 1, as2, ad2, m2, ex2, den2);
    k_agg2<<<cdiv((long)nE * 32, 256), 256, 0, stream>>>(ei, h2, ex2, den2, agg2);

    // ---- output ----
    k_logsoftmax<<<cdiv(NN, 256), 256, 0, stream>>>(agg2, b2, out);
}
